// DegModel_2637109920615
// MI455X (gfx1250) — matrix operand order repaired and checked by reading, not compiled, not run on hardware
//
#include <hip/hip_runtime.h>
#include <math.h>

// ---------------------------------------------------------------------------
// Types for CDNA5 WMMA (gfx1250, wave32)
// ---------------------------------------------------------------------------
typedef __attribute__((ext_vector_type(16))) __bf16 v16bf;
typedef __attribute__((ext_vector_type(8)))  float  v8f;

#if __has_builtin(__builtin_amdgcn_sched_barrier)
#define SCHED_BARRIER() __builtin_amdgcn_sched_barrier(0)
#else
#define SCHED_BARRIER()
#endif

__device__ __forceinline__ unsigned short f2bf(float f) {
    unsigned u = __float_as_uint(f);
    unsigned r = (u + 0x7FFFu + ((u >> 16) & 1u)) >> 16;   // RNE
    return (unsigned short)r;
}

// Padded NHWC activation geometry: [B][130][130][64] bf16, halo ring = 0
#define PH 130
#define PW 130
#define PC 64
#define PADDED_N (8 * PH * PW * PC)

// ---------------------------------------------------------------------------
// Parameter tables (device pointers passed by value in kernarg)
// ---------------------------------------------------------------------------
struct StackParams {
    const float *hw, *hb, *hg, *hbe, *tw, *tb;
    const float *w1[8], *b1[8], *g[8], *be[8], *w2[8], *b2[8];
};
struct W16 { const float* w[16]; };

// ---------------------------------------------------------------------------
// 1) KernelModel: zk(8,64) -> MLP(BN over batch) -> softmax(441). One block.
// ---------------------------------------------------------------------------
__global__ void __launch_bounds__(512)
kernel_model_kernel(const float* __restrict__ zk, StackParams kp,
                    float* __restrict__ kvec, float* __restrict__ kout) {
    __shared__ float zs[512], ys[512], rs[512], mbuf[64], vbuf[64], klog[3528];
    int tid = threadIdx.x;
    zs[tid] = zk[tid];
    __syncthreads();
    int b = tid >> 6, o = tid & 63;

    // head 1x1 conv
    {
        float a = kp.hb[o];
        const float* w = kp.hw + o * 64;
        const float* z = zs + b * 64;
        for (int i = 0; i < 64; ++i) a += w[i] * z[i];
        ys[tid] = a;
    }
    __syncthreads();
    if (tid < 64) {
        float s = 0.f, q = 0.f;
        for (int bb = 0; bb < 8; ++bb) { float v = ys[bb * 64 + tid]; s += v; q += v * v; }
        float m = s * 0.125f; mbuf[tid] = m; vbuf[tid] = q * 0.125f - m * m;
    }
    __syncthreads();
    ys[tid] = fmaxf(kp.hg[o] * (ys[tid] - mbuf[o]) * rsqrtf(vbuf[o] + 1e-5f) + kp.hbe[o], 0.f);
    __syncthreads();

    for (int k = 0; k < 8; ++k) {
        float r = kp.b1[k][o];
        { const float* w = kp.w1[k] + o * 64; const float* y = ys + b * 64;
          for (int i = 0; i < 64; ++i) r += w[i] * y[i]; }
        rs[tid] = r;
        __syncthreads();
        if (tid < 64) {
            float s = 0.f, q = 0.f;
            for (int bb = 0; bb < 8; ++bb) { float v = rs[bb * 64 + tid]; s += v; q += v * v; }
            float m = s * 0.125f; mbuf[tid] = m; vbuf[tid] = q * 0.125f - m * m;
        }
        __syncthreads();
        rs[tid] = fmaxf(kp.g[k][o] * (rs[tid] - mbuf[o]) * rsqrtf(vbuf[o] + 1e-5f) + kp.be[k][o], 0.f);
        __syncthreads();
        float r2 = kp.b2[k][o];
        { const float* w = kp.w2[k] + o * 64; const float* rr = rs + b * 64;
          for (int i = 0; i < 64; ++i) r2 += w[i] * rr[i]; }
        __syncthreads();
        ys[tid] += r2;
        __syncthreads();
    }

    // tail 64 -> 441
    for (int idx = tid; idx < 3528; idx += 512) {
        int bb = idx / 441, q = idx % 441;
        float a = kp.tb[q];
        const float* w = kp.tw + q * 64;
        const float* y = ys + bb * 64;
        for (int i = 0; i < 64; ++i) a += w[i] * y[i];
        klog[idx] = a;
    }
    __syncthreads();
    if (tid < 8) {
        float* kl = klog + tid * 441;
        float mx = -1e30f;
        for (int q = 0; q < 441; ++q) mx = fmaxf(mx, kl[q]);
        float s = 0.f;
        for (int q = 0; q < 441; ++q) { float e = expf(kl[q] - mx); kl[q] = e; s += e; }
        float inv = 1.f / s;
        for (int q = 0; q < 441; ++q) kl[q] *= inv;
    }
    __syncthreads();
    for (int idx = tid; idx < 3528; idx += 512) { kvec[idx] = klog[idx]; kout[idx] = klog[idx]; }
}

// ---------------------------------------------------------------------------
// 2) Blur + stride-4 downsample with reflect pad. 16x16 output tile per block.
// ---------------------------------------------------------------------------
__global__ void __launch_bounds__(256)
blur_kernel(const float* __restrict__ inp, const float* __restrict__ kvec,
            float* __restrict__ xlr) {
    __shared__ float patch[81 * 84];
    __shared__ float kv[441];
    int bc = blockIdx.z;                  // b*3 + c
    int b  = bc / 3;
    int i0 = blockIdx.y * 16, j0 = blockIdx.x * 16;
    int tid = threadIdx.x;
    for (int t = tid; t < 441; t += 256) kv[t] = kvec[b * 441 + t];
    const float* img = inp + (size_t)bc * 512 * 512;
    int rbase = 4 * i0 - 10, cbase = 4 * j0 - 10;
    for (int idx = tid; idx < 81 * 81; idx += 256) {
        int rr = idx / 81, cc = idx % 81;
        int r = rbase + rr; r = (r < 0) ? -r : r; r = (r > 511) ? 1022 - r : r;
        int c = cbase + cc; c = (c < 0) ? -c : c; c = (c > 511) ? 1022 - c : c;
        patch[rr * 84 + cc] = img[r * 512 + c];
    }
    __syncthreads();
    int ty = tid >> 4, tx = tid & 15;
    float acc = 0.f;
    const float* p0 = patch + (4 * ty) * 84 + 4 * tx;
    for (int ky = 0; ky < 21; ++ky) {
        const float* pr = p0 + ky * 84;
        const float* kr = kv + ky * 21;
        for (int kx = 0; kx < 21; ++kx) acc += kr[kx] * pr[kx];
    }
    xlr[(size_t)bc * 16384 + (i0 + ty) * 128 + (j0 + tx)] = acc;
}

// ---------------------------------------------------------------------------
// 3) Weight pre-pack: OIHW fp32 -> per-lane A-matrix bf16 layout
//    (contiguous K order): lane<16 holds K {0..15},
//    lane>=16 holds K {16..31} for its M = lane%16 row.
// ---------------------------------------------------------------------------
__global__ void __launch_bounds__(256)
prepack_kernel(W16 wp, unsigned short* __restrict__ apack) {
    int ci  = blockIdx.y;
    int idx = blockIdx.x * 256 + threadIdx.x;      // < 36864
    int e = idx & 15, lane = (idx >> 4) & 31, mt = (idx >> 9) & 3;
    int kc = (idx >> 11) & 1, tap = idx >> 12;
    int o   = mt * 16 + (lane & 15);
    int off = (lane < 16) ? 0 : 16;
    int k = off + e;
    int i = kc * 32 + k;
    int ky = tap / 3, kx = tap % 3;
    float v = wp.w[ci][((o * 64 + i) * 3 + ky) * 3 + kx];
    apack[(size_t)ci * 36864 + idx] = f2bf(v);
}

// ---------------------------------------------------------------------------
// 4) 3x3 conv 64->64 as implicit GEMM with v_wmma_f32_16x16x32_bf16.
//    Input activations in zero-padded NHWC bf16 [8][130][130][64]: each
//    B-operand (32 K-channels x 16 pixels) is a direct 32B-aligned global
//    load per lane. Software-pipelined 2 groups deep; sched_barrier(0) pins
//    the prefetch loads above the WMMAs so the scheduler cannot sink them,
//    keeping ~2 operand-groups of loads in flight behind partial waits.
//    Block = 128 threads (4 waves); tile = 64 out-ch x 64 pixels of one row.
// ---------------------------------------------------------------------------
__device__ __forceinline__ const v16bf* Ap(const unsigned short* abase, int g) {
    // apack group stride: 128 lanes * 16 elems = 2048 ushorts
    return reinterpret_cast<const v16bf*>(abase + (size_t)g * 2048);
}
__device__ __forceinline__ const v16bf* Bp(const unsigned short* bbase, int g, int ns) {
    int dy = g / 6, rem = g % 6, dx = rem >> 1, kc = rem & 1;
    return reinterpret_cast<const v16bf*>(
        bbase + (size_t)dy * (PW * PC) + (size_t)(dx + (ns << 4)) * PC + (kc << 5));
}

__global__ void __launch_bounds__(128)
conv3x3_wmma_kernel(const unsigned short* __restrict__ inbf,
                    const unsigned short* __restrict__ apack,
                    const float* __restrict__ bias,
                    float* __restrict__ out, float* stats, int do_stats) {
    __shared__ float ssum[64], ssq[64];
    int tid = threadIdx.x;
    int x0 = blockIdx.x * 64, y = blockIdx.y, b = blockIdx.z;
    int wv = tid >> 5, lane = tid & 31;
    int half = lane >> 4, ln = lane & 15;
    int obase = wv << 4;

    if (do_stats) {
        if (tid < 64) { ssum[tid] = 0.f; ssq[tid] = 0.f; }
        __syncthreads();
    }

    v8f acc[4];
    for (int r = 0; r < 8; ++r) {
        float bv = bias[obase + (half << 3) + r];
        acc[0][r] = bv; acc[1][r] = bv; acc[2][r] = bv; acc[3][r] = bv;
    }

    // Per-lane base pointers.
    const unsigned short* abase = apack + ((size_t)(wv * 32 + lane) << 4);
    const unsigned short* bbase = inbf + (((size_t)b * PH + y) * PW + x0 + ln) * PC
                                       + (half << 4);

    // Prologue: groups 0 and 1 in flight.
    v16bf aC = *Ap(abase, 0), aN = *Ap(abase, 1);
    v16bf b0C = *Bp(bbase, 0, 0), b1C = *Bp(bbase, 0, 1),
          b2C = *Bp(bbase, 0, 2), b3C = *Bp(bbase, 0, 3);
    v16bf b0N = *Bp(bbase, 1, 0), b1N = *Bp(bbase, 1, 1),
          b2N = *Bp(bbase, 1, 2), b3N = *Bp(bbase, 1, 3);

    #pragma unroll
    for (int g = 0; g < 18; ++g) {
        v16bf aF{}, b0F{}, b1F{}, b2F{}, b3F{};
        if (g + 2 < 18) {
            aF  = *Ap(abase, g + 2);
            b0F = *Bp(bbase, g + 2, 0);
            b1F = *Bp(bbase, g + 2, 1);
            b2F = *Bp(bbase, g + 2, 2);
            b3F = *Bp(bbase, g + 2, 3);
        }
        // Pin: prefetch loads above may not be sunk below; WMMAs below may not
        // be hoisted above. Keeps 2 operand groups of loads in flight.
        SCHED_BARRIER();
        acc[0] = __builtin_amdgcn_wmma_f32_16x16x32_bf16(
            false, aC, false, b0C, (short)0, acc[0], false, false);
        acc[1] = __builtin_amdgcn_wmma_f32_16x16x32_bf16(
            false, aC, false, b1C, (short)0, acc[1], false, false);
        acc[2] = __builtin_amdgcn_wmma_f32_16x16x32_bf16(
            false, aC, false, b2C, (short)0, acc[2], false, false);
        acc[3] = __builtin_amdgcn_wmma_f32_16x16x32_bf16(
            false, aC, false, b3C, (short)0, acc[3], false, false);
        aC = aN; b0C = b0N; b1C = b1N; b2C = b2N; b3C = b3N;
        aN = aF; b0N = b0F; b1N = b1F; b2N = b2F; b3N = b3F;
    }

    // Store fp32 conv output (pre-BN), NCHW.
    float* orow = out + (((size_t)b << 6) << 14) + (y << 7);
    #pragma unroll
    for (int ns = 0; ns < 4; ++ns) {
        int x = x0 + (ns << 4) + ln;
        #pragma unroll
        for (int r = 0; r < 8; ++r) {
            int o = obase + (half << 3) + r;
            orow[((size_t)o << 14) + x] = acc[ns][r];
        }
    }

    // BN batch-statistics accumulation: per-channel sum / sumsq.
    if (do_stats) {
        for (int r = 0; r < 8; ++r) {
            float s = 0.f, q = 0.f;
            #pragma unroll
            for (int ns = 0; ns < 4; ++ns) { float v = acc[ns][r]; s += v; q += v * v; }
            s += __shfl_xor(s, 1, 16); s += __shfl_xor(s, 2, 16);
            s += __shfl_xor(s, 4, 16); s += __shfl_xor(s, 8, 16);
            q += __shfl_xor(q, 1, 16); q += __shfl_xor(q, 2, 16);
            q += __shfl_xor(q, 4, 16); q += __shfl_xor(q, 8, 16);
            if (ln == 0) {
                int o = obase + (half << 3) + r;
                atomicAdd(&ssum[o], s); atomicAdd(&ssq[o], q);
            }
        }
        __syncthreads();
        if (tid < 64) { atomicAdd(&stats[tid], ssum[tid]); atomicAdd(&stats[64 + tid], ssq[tid]); }
    }
}

// ---------------------------------------------------------------------------
// 5) Head conv (11->64, 3x3, pad1) scalar, with fused stats (block = one (b,o)).
// ---------------------------------------------------------------------------
__global__ void __launch_bounds__(256)
head_conv_kernel(const float* __restrict__ xlr, const float* __restrict__ zn,
                 const float* __restrict__ w, const float* __restrict__ bias,
                 float* __restrict__ out, float* stats) {
    int idx = blockIdx.x * 256 + threadIdx.x;
    int x = idx & 127, yy = (idx >> 7) & 127, o = (idx >> 14) & 63, b = idx >> 20;
    float acc = bias[o];
    for (int c = 0; c < 11; ++c) {
        const float* img = (c < 3) ? (xlr + ((size_t)(b * 3 + c) << 14))
                                   : (zn + ((size_t)(b * 8 + c - 3) << 14));
        const float* wc = w + (o * 11 + c) * 9;
        for (int dy = 0; dy < 3; ++dy) {
            int yi = yy + dy - 1; if (yi < 0 || yi > 127) continue;
            for (int dx = 0; dx < 3; ++dx) {
                int xi = x + dx - 1; if (xi < 0 || xi > 127) continue;
                acc += wc[dy * 3 + dx] * img[(yi << 7) + xi];
            }
        }
    }
    out[idx] = acc;
    __shared__ float s1[256], s2[256];
    s1[threadIdx.x] = acc; s2[threadIdx.x] = acc * acc;
    __syncthreads();
    for (int st = 128; st > 0; st >>= 1) {
        if (threadIdx.x < st) { s1[threadIdx.x] += s1[threadIdx.x + st];
                                s2[threadIdx.x] += s2[threadIdx.x + st]; }
        __syncthreads();
    }
    if (threadIdx.x == 0) { atomicAdd(&stats[o], s1[0]); atomicAdd(&stats[64 + o], s2[0]); }
}

// ---------------------------------------------------------------------------
// 6) Elementwise BN (train-mode, biased var, N=131072) + ReLU.
//    Writes bf16 into the padded NHWC buffer (+ optional fp32 NCHW master).
// ---------------------------------------------------------------------------
__global__ void __launch_bounds__(256)
bnrelu_kernel(const float* __restrict__ t, const float* __restrict__ stats,
              const float* __restrict__ g, const float* __restrict__ be,
              unsigned short* __restrict__ obf, float* __restrict__ of32, int wf32) {
    int idx = blockIdx.x * 256 + threadIdx.x;
    int x = idx & 127, yy = (idx >> 7) & 127, ch = (idx >> 14) & 63, b = idx >> 20;
    const float invN = 1.f / 131072.f;
    float m = stats[ch] * invN;
    float v = stats[64 + ch] * invN - m * m;
    float val = g[ch] * (t[idx] - m) * rsqrtf(v + 1e-5f) + be[ch];
    val = fmaxf(val, 0.f);
    obf[(((size_t)b * PH + yy + 1) * PW + x + 1) * PC + ch] = f2bf(val);
    if (wf32) of32[idx] = val;
}

// ---------------------------------------------------------------------------
// 7) Residual add: y += t; refresh padded-NHWC bf16 copy.
// ---------------------------------------------------------------------------
__global__ void __launch_bounds__(256)
add_kernel(float* __restrict__ y, const float* __restrict__ t,
           unsigned short* __restrict__ ybf) {
    int idx = blockIdx.x * 256 + threadIdx.x;
    int x = idx & 127, yy = (idx >> 7) & 127, ch = (idx >> 14) & 63, b = idx >> 20;
    float v = y[idx] + t[idx];
    y[idx] = v;
    ybf[(((size_t)b * PH + yy + 1) * PW + x + 1) * PC + ch] = f2bf(v);
}

// ---------------------------------------------------------------------------
// 8) Tail 1x1 conv 64->3 (= noise) and final out0 = x_lr + noise.
// ---------------------------------------------------------------------------
__global__ void __launch_bounds__(256)
tail_kernel(const float* __restrict__ y, const float* __restrict__ tw,
            const float* __restrict__ tb, const float* __restrict__ xlr,
            float* __restrict__ noise, float* __restrict__ out0) {
    int idx = blockIdx.x * 256 + threadIdx.x;           // < 393216
    int x = idx & 127, yy = (idx >> 7) & 127;
    int c = (idx >> 14) % 3, b = idx / (3 << 14);
    const float* yp = y + ((size_t)b << 20) + (yy << 7) + x;
    float acc = tb[c];
    const float* twc = tw + c * 64;
    for (int o = 0; o < 64; ++o) acc += twc[o] * yp[(size_t)o << 14];
    noise[idx] = acc;
    out0[idx] = xlr[idx] + acc;
}

__global__ void zero_kernel(float* p, int n) {
    int i = blockIdx.x * 256 + threadIdx.x;
    if (i < n) p[i] = 0.f;
}

__global__ void zero_bf_kernel(unsigned short* p, int n) {
    int i = blockIdx.x * 256 + threadIdx.x;
    if (i < n) p[i] = 0;
}

// ---------------------------------------------------------------------------
// Host orchestration
// ---------------------------------------------------------------------------
static void fill_stack(StackParams& s, void* const* din, int base, bool sortedNested) {
    auto F = [&](int i) { return (const float*)din[base + i]; };
    if (sortedNested) {  // tree_leaves: blocks[k]{b1,b2,be,g,w1,w2} then hb,hbe,hg,hw,tb,tw
        for (int k = 0; k < 8; ++k) {
            s.b1[k] = F(6 * k + 0); s.b2[k] = F(6 * k + 1); s.be[k] = F(6 * k + 2);
            s.g[k]  = F(6 * k + 3); s.w1[k] = F(6 * k + 4); s.w2[k] = F(6 * k + 5);
        }
        s.hb = F(48); s.hbe = F(49); s.hg = F(50); s.hw = F(51); s.tb = F(52); s.tw = F(53);
    } else {             // insertion order: hw,hb,hg,hbe,tw,tb then blocks{w1,b1,g,be,w2,b2}
        s.hw = F(0); s.hb = F(1); s.hg = F(2); s.hbe = F(3); s.tw = F(4); s.tb = F(5);
        for (int k = 0; k < 8; ++k) {
            s.w1[k] = F(6 + 6 * k); s.b1[k] = F(7 + 6 * k); s.g[k]  = F(8 + 6 * k);
            s.be[k] = F(9 + 6 * k); s.w2[k] = F(10 + 6 * k); s.b2[k] = F(11 + 6 * k);
        }
    }
}

extern "C" void kernel_launch(void* const* d_in, const int* in_sizes, int n_in,
                              void* d_out, int out_size, void* d_ws, size_t ws_size,
                              hipStream_t stream) {
    if (n_in < 111) return;

    int i_inp = 0, i_zk, i_zn, kbase, nbase; bool sortedNested;
    if (in_sizes[1] == 512) {                 // top-level insertion order
        i_zk = 1; i_zn = 2; kbase = 3; nbase = 57;
        sortedNested = (in_sizes[3] == 64);   // sorted leaves start with blocks[0].b1 (64)
    } else {                                  // fully-sorted pytree flattening
        kbase = 1; nbase = 55; i_zk = 109; i_zn = 110; sortedNested = true;
    }

    StackParams kp, np;
    fill_stack(kp, d_in, kbase, sortedNested);
    fill_stack(np, d_in, nbase, sortedNested);

    const float* inp = (const float*)d_in[i_inp];
    const float* zk  = (const float*)d_in[i_zk];
    const float* zn  = (const float*)d_in[i_zn];

    // Workspace layout (all offsets multiples of 256)
    char* ws = (char*)d_ws;
    size_t off = 0;
    float* kvec = (float*)(ws + off);           off += 14336;        // 8*441 f32
    float* xlr  = (float*)(ws + off);           off += 1572864;      // 8*3*128*128 f32
    float* yf   = (float*)(ws + off);           off += 33554432;     // 8*64*128*128 f32
    float* tf   = (float*)(ws + off);           off += 33554432;
    unsigned short* ybf = (unsigned short*)(ws + off); off += (size_t)PADDED_N * 2; // 17.3MB
    unsigned short* rbf = (unsigned short*)(ws + off); off += (size_t)PADDED_N * 2;
    float* stats = (float*)(ws + off);          off += 512;          // 2*64 f32
    unsigned short* apack = (unsigned short*)(ws + off); off += 16 * 36864 * 2;
    (void)ws_size;

    float* out0 = (float*)d_out;
    float* outK = out0 + 393216;
    float* outN = out0 + 393216 + 3528;

    // --- zero padded NHWC buffers once (halo ring must be 0 for all convs) ---
    int zgrid = (PADDED_N + 255) / 256;
    zero_bf_kernel<<<zgrid, 256, 0, stream>>>(ybf, PADDED_N);
    zero_bf_kernel<<<zgrid, 256, 0, stream>>>(rbf, PADDED_N);

    // --- kernel model + blur ---
    kernel_model_kernel<<<1, 512, 0, stream>>>(zk, kp, kvec, outK);
    blur_kernel<<<dim3(8, 8, 24), 256, 0, stream>>>(inp, kvec, xlr);

    // --- pre-pack all 16 resblock conv weights into WMMA A-layout bf16 ---
    W16 wp;
    for (int k = 0; k < 8; ++k) { wp.w[2 * k] = np.w1[k]; wp.w[2 * k + 1] = np.w2[k]; }
    prepack_kernel<<<dim3(144, 16), 256, 0, stream>>>(wp, apack);

    // --- head conv + BN/ReLU ---
    zero_kernel<<<1, 128, 0, stream>>>(stats, 128);
    head_conv_kernel<<<32768, 256, 0, stream>>>(xlr, zn, np.hw, np.hb, tf, stats);
    bnrelu_kernel<<<32768, 256, 0, stream>>>(tf, stats, np.hg, np.hbe, ybf, yf, 1);

    // --- 8 resblocks: conv1(WMMA,+stats) -> BN/ReLU -> conv2(WMMA) -> add ---
    for (int k = 0; k < 8; ++k) {
        zero_kernel<<<1, 128, 0, stream>>>(stats, 128);
        conv3x3_wmma_kernel<<<dim3(2, 128, 8), 128, 0, stream>>>(
            ybf, apack + (size_t)(2 * k) * 36864, np.b1[k], tf, stats, 1);
        bnrelu_kernel<<<32768, 256, 0, stream>>>(tf, stats, np.g[k], np.be[k], rbf, yf, 0);
        conv3x3_wmma_kernel<<<dim3(2, 128, 8), 128, 0, stream>>>(
            rbf, apack + (size_t)(2 * k + 1) * 36864, np.b2[k], tf, stats, 0);
        add_kernel<<<32768, 256, 0, stream>>>(yf, tf, ybf);
    }

    // --- tail 1x1 conv -> noise, out0 = x_lr + noise ---
    tail_kernel<<<1536, 256, 0, stream>>>(yf, np.tw, np.tb, xlr, outN, out0);
    (void)out_size;
}
